// GATv2Layer_89146341195955
// MI455X (gfx1250) — matrix operand order repaired and checked by reading, not compiled, not run on hardware
//
#include <hip/hip_runtime.h>
#include <math.h>

// ---------------------------------------------------------------------------
// GATv2 layer for MI455X (gfx1250, wave32).
//   xl = x@W_l + b_l ; xr = x@W_r + b_r      -> dual f16 WMMA GEMM (f32 accum)
//   per-edge attention logits + segment softmax over dst (atomic max / add)
//   out = LN(elu(segsum(alpha*xl[src]) + bias) + x)
// ---------------------------------------------------------------------------

typedef __attribute__((ext_vector_type(16))) _Float16 v16h;
typedef __attribute__((ext_vector_type(8)))  float    v8f;

#define INC  128
#define OUTC 128
#define HH   8
#define DD   16
#define NEG_SLOPE 0.2f

union F16U { _Float16 h; unsigned short u; };

__device__ __forceinline__ unsigned short f2h_bits(float f) {
  F16U x; x.h = (_Float16)f; return x.u;
}

// monotone order-preserving float<->uint encoding for atomicMax on floats
__device__ __forceinline__ unsigned encf(float f) {
  unsigned u = __float_as_uint(f);
  return (u & 0x80000000u) ? ~u : (u | 0x80000000u);
}
__device__ __forceinline__ float decf(unsigned u) {
  unsigned v = (u & 0x80000000u) ? (u & 0x7FFFFFFFu) : ~u;
  return __uint_as_float(v);
}

// ---------------------------------------------------------------------------
// Kernel 1: pre-swizzle W_l / W_r (f32, [K=128][N=128] row-major) into the
// per-lane B-fragment layout of V_WMMA_F32_16X16X32_F16:
//   B is 32x16 (KxN) per k-tile; lane l holds column N = l&15;
//   lanes 0-15 hold K 0-7,16-23, lanes 16-31 hold K 8-15,24-31; dword v = {K=kb+2v', K=kb+2v'+1}.
// Layout in ws: pack[((kt*8 + nt)*32 + lane)*8 + v]  (uint32, lo=even K)
// total: 2 matrices * 4 ktiles * 8 ntiles * 32 lanes * 8 dwords = 16384 dwords
// ---------------------------------------------------------------------------
__global__ void gat_pack_weights(const float* __restrict__ Wl,
                                 const float* __restrict__ Wr,
                                 unsigned* __restrict__ Blp,
                                 unsigned* __restrict__ Brp) {
  int i = blockIdx.x * blockDim.x + threadIdx.x;
  if (i >= 2 * 8192) return;
  int which = i >> 13;
  int r     = i & 8191;
  int v     = r & 7;
  int lane  = (r >> 3) & 31;
  int nt    = (r >> 8) & 7;
  int kt    = (r >> 11) & 3;
  int n  = nt * 16 + (lane & 15);
  int k0 = kt * 32 + ((lane < 16) ? 0 : 8) + 2 * v + ((v >= 4) ? 8 : 0);
  const float* W = which ? Wr : Wl;
  unsigned lo = f2h_bits(W[(size_t)k0 * OUTC + n]);
  unsigned hi = f2h_bits(W[(size_t)(k0 + 1) * OUTC + n]);
  (which ? Brp : Blp)[r] = lo | (hi << 16);
}

// ---------------------------------------------------------------------------
// Kernel 2: dual GEMM. One wave per 16x16 output tile; 8 waves/block cover the
// 8 column tiles of one 16-row stripe. A fragment (16x32 f16) loaded from x:
//   lane l holds row M = l&15; K offsets {kb..kb+7} u {kb+16..kb+23},
//   kb = (l<16)?0:8  (per ISA 16-bit A-matrix table). A is shared between the
// W_l and W_r products -> 2 WMMAs per k-step, 8 WMMAs total per wave.
// ---------------------------------------------------------------------------
__global__ void __launch_bounds__(256)
gat_dual_gemm(const float* __restrict__ x,
              const unsigned* __restrict__ Blp,
              const unsigned* __restrict__ Brp,
              const float* __restrict__ bl_bias,
              const float* __restrict__ br_bias,
              float* __restrict__ xl,
              float* __restrict__ xr,
              int nrows) {
  const int lane  = threadIdx.x & 31;
  const int nt    = threadIdx.x >> 5;        // wave id == column tile 0..7
  const int mtile = blockIdx.x;
  int row = mtile * 16 + (lane & 15);
  if (row >= nrows) row = nrows - 1;         // clamp (keeps EXEC all-ones)
  const int kb = (lane < 16) ? 0 : 8;

  v8f cl = {};
  v8f cr = {};
#pragma unroll
  for (int kt = 0; kt < 4; ++kt) {
    const float* rp = x + (size_t)row * INC + kt * 32 + kb;
    float4 f0 = *(const float4*)(rp);
    float4 f1 = *(const float4*)(rp + 4);
    float4 f2 = *(const float4*)(rp + 16);
    float4 f3 = *(const float4*)(rp + 20);
    v16h a;
    a[0]  = (_Float16)f0.x; a[1]  = (_Float16)f0.y; a[2]  = (_Float16)f0.z; a[3]  = (_Float16)f0.w;
    a[4]  = (_Float16)f1.x; a[5]  = (_Float16)f1.y; a[6]  = (_Float16)f1.z; a[7]  = (_Float16)f1.w;
    a[8]  = (_Float16)f2.x; a[9]  = (_Float16)f2.y; a[10] = (_Float16)f2.z; a[11] = (_Float16)f2.w;
    a[12] = (_Float16)f3.x; a[13] = (_Float16)f3.y; a[14] = (_Float16)f3.z; a[15] = (_Float16)f3.w;

    size_t boff = ((size_t)(kt * 8 + nt) * 32 + lane) * 8;
    v16h bl = *(const v16h*)(Blp + boff);
    v16h br = *(const v16h*)(Brp + boff);

    cl = __builtin_amdgcn_wmma_f32_16x16x32_f16(false, a, false, bl, (short)0, cl, false, false);
    cr = __builtin_amdgcn_wmma_f32_16x16x32_f16(false, a, false, br, (short)0, cr, false, false);
  }

  // C/D layout: vgpr r, lanes 0-15 -> M=r, N=lane; lanes 16-31 -> M=8+r
  const int ncol  = nt * 16 + (lane & 15);
  const int mbase = mtile * 16 + ((lane < 16) ? 0 : 8);
  const float bb_l = bl_bias[ncol];
  const float bb_r = br_bias[ncol];
#pragma unroll
  for (int r = 0; r < 8; ++r) {
    int m = mbase + r;
    if (m < nrows) {
      size_t o = (size_t)m * OUTC + ncol;
      xl[o] = cl[r] + bb_l;
      xr[o] = cr[r] + bb_r;
    }
  }
}

// ---------------------------------------------------------------------------
// Kernel 3: per-edge attention logits + segment max.
// 128 threads per edge (one per feature); head h = t>>4. 16-lane shuffle
// reduction of m*att per head; lane (t&15)==0 stores logit and atomicMax.
// ---------------------------------------------------------------------------
__global__ void __launch_bounds__(256)
gat_edge_logits(const int* __restrict__ ei,
                const float* __restrict__ xl,
                const float* __restrict__ xr,
                const float* __restrict__ att,
                float* __restrict__ elog,
                unsigned* __restrict__ emaxk,
                int E, long long Etot) {
  long long e = (long long)blockIdx.x * 2 + (threadIdx.x >> 7);
  if (e >= Etot) return;
  const int t = threadIdx.x & 127;
  const int src = (e < E) ? ei[e] : (int)(e - E);
  const int dst = (e < E) ? ei[(size_t)E + e] : (int)(e - E);

  float m = xl[(size_t)src * OUTC + t] + xr[(size_t)dst * OUTC + t];
  m = (m >= 0.f) ? m : NEG_SLOPE * m;
  float p = m * att[t];                 // att flat [H*D] == feature index t
  p += __shfl_xor(p, 1, 32);
  p += __shfl_xor(p, 2, 32);
  p += __shfl_xor(p, 4, 32);
  p += __shfl_xor(p, 8, 32);
  if ((t & 15) == 0) {
    int h = t >> 4;
    elog[e * HH + h] = p;
    atomicMax(&emaxk[(size_t)dst * HH + h], encf(p));
  }
}

// ---------------------------------------------------------------------------
// Kernel 4: exp(e - max), scatter-add weighted source features + denominators.
// ---------------------------------------------------------------------------
__global__ void __launch_bounds__(256)
gat_edge_scatter(const int* __restrict__ ei,
                 const float* __restrict__ xl,
                 const float* __restrict__ elog,
                 const unsigned* __restrict__ emaxk,
                 float* __restrict__ outacc,
                 float* __restrict__ denom,
                 int E, long long Etot) {
  long long e = (long long)blockIdx.x * 2 + (threadIdx.x >> 7);
  if (e >= Etot) return;
  const int t = threadIdx.x & 127;
  const int h = t >> 4;
  const int src = (e < E) ? ei[e] : (int)(e - E);
  const int dst = (e < E) ? ei[(size_t)E + e] : (int)(e - E);

  float ex = expf(elog[e * HH + h] - decf(emaxk[(size_t)dst * HH + h]));
  atomicAdd(&outacc[(size_t)dst * OUTC + t], ex * xl[(size_t)src * OUTC + t]);
  if ((t & 15) == 0) atomicAdd(&denom[(size_t)dst * HH + h], ex);
}

// ---------------------------------------------------------------------------
// Kernel 5: normalize, bias, ELU, residual, LayerNorm. One node per 128-thread
// block (4 waves); wave shuffle + LDS reduction for mean/var.
// ---------------------------------------------------------------------------
__global__ void __launch_bounds__(128)
gat_finalize(const float* __restrict__ outacc,
             const float* __restrict__ denom,
             const float* __restrict__ bias,
             const float* __restrict__ x,
             const float* __restrict__ gamma,
             const float* __restrict__ beta,
             float* __restrict__ out) {
  const int i = blockIdx.x;
  const int t = threadIdx.x;
  float raw = outacc[(size_t)i * OUTC + t] /
              (denom[(size_t)i * HH + (t >> 4)] + 1e-16f) + bias[t];
  float v = (raw > 0.f) ? raw : expm1f(raw);   // ELU
  v += x[(size_t)i * OUTC + t];                // residual

  float s = v, q = v * v;
  for (int m = 16; m >= 1; m >>= 1) {
    s += __shfl_xor(s, m, 32);
    q += __shfl_xor(q, m, 32);
  }
  __shared__ float ss[4], sq[4];
  const int wave = t >> 5, lane = t & 31;
  if (lane == 0) { ss[wave] = s; sq[wave] = q; }
  __syncthreads();
  float tot = ss[0] + ss[1] + ss[2] + ss[3];
  float tq  = sq[0] + sq[1] + sq[2] + sq[3];
  float mu  = tot * (1.f / 128.f);
  float var = tq * (1.f / 128.f) - mu * mu;
  out[(size_t)i * OUTC + t] = (v - mu) * rsqrtf(var + 1e-5f) * gamma[t] + beta[t];
}

// ---------------------------------------------------------------------------
extern "C" void kernel_launch(void* const* d_in, const int* in_sizes, int n_in,
                              void* d_out, int out_size, void* d_ws, size_t ws_size,
                              hipStream_t stream) {
  const float* x     = (const float*)d_in[0];
  const int*   ei    = (const int*)d_in[1];
  const float* Wl    = (const float*)d_in[2];
  const float* bl    = (const float*)d_in[3];
  const float* Wr    = (const float*)d_in[4];
  const float* br    = (const float*)d_in[5];
  const float* att   = (const float*)d_in[6];
  const float* bias  = (const float*)d_in[7];
  const float* gam   = (const float*)d_in[8];
  const float* bet   = (const float*)d_in[9];
  float* out = (float*)d_out;

  const int       Nn   = in_sizes[0] / INC;
  const int       E    = in_sizes[1] / 2;
  const long long Etot = (long long)E + Nn;

  // workspace carve-up (all chunks 32B aligned; ~189 MB for N=1e5, E=8e5)
  float* ws = (float*)d_ws;
  float*    xl     = ws;            ws += (size_t)Nn * OUTC;
  float*    xrw    = ws;            ws += (size_t)Nn * OUTC;
  float*    outacc = ws;            ws += (size_t)Nn * OUTC;
  float*    denom  = ws;            ws += (size_t)Nn * HH;
  unsigned* emaxk  = (unsigned*)ws; ws += (size_t)Nn * HH;
  float*    elog   = ws;            ws += (size_t)Etot * HH;
  unsigned* Blp    = (unsigned*)ws; ws += 8192;
  unsigned* Brp    = (unsigned*)ws; ws += 8192;

  hipMemsetAsync(outacc, 0, (size_t)Nn * OUTC * sizeof(float), stream);
  hipMemsetAsync(denom,  0, (size_t)Nn * HH * sizeof(float), stream);
  hipMemsetAsync(emaxk,  0, (size_t)Nn * HH * sizeof(unsigned), stream);

  gat_pack_weights<<<(16384 + 255) / 256, 256, 0, stream>>>(Wl, Wr, Blp, Brp);

  const int mtiles = (Nn + 15) / 16;
  gat_dual_gemm<<<mtiles, 256, 0, stream>>>(x, Blp, Brp, bl, br, xl, xrw, Nn);

  const unsigned eblocks = (unsigned)((Etot + 1) / 2);
  gat_edge_logits<<<eblocks, 256, 0, stream>>>(ei, xl, xrw, att, elog, emaxk, E, Etot);
  gat_edge_scatter<<<eblocks, 256, 0, stream>>>(ei, xl, elog, emaxk, outacc, denom, E, Etot);

  gat_finalize<<<Nn, 128, 0, stream>>>(outacc, denom, bias, x, gam, bet, out);
}
